// HierarchicalAutoregressivePolicy_90391881712216
// MI455X (gfx1250) — matrix operand order repaired and checked by reading, not compiled, not run on hardware
//
#include <hip/hip_runtime.h>
#include <math.h>

// ---------------------------------------------------------------------------
// HierarchicalAutoregressivePolicy on MI455X (gfx1250), bf16-WMMA everywhere.
// Weights stay fp32 in HBM (read once, ~590MB -> ~25us @ 23.3TB/s floor),
// converted to bf16 on the fly in LDS; accumulation in fp32 via
// v_wmma_f32_16x16x32_bf16. Attention is exact-causal (the reference's block
// mask reduces to k>q). Padded rows (307..319) are zeroed after each LN.
// ---------------------------------------------------------------------------

#define NB_B   32
#define SEQP   320          // padded sequence (20 tiles of 16)
#define SEQR   307          // real sequence
#define DM     1024
#define NH     16
#define DHD    64
#define DFF_   4096
#define NLAY   12
#define DIMG   768
#define DLG    512
#define GS0    259          // goal token start (== HIST)
#define MTOT   (NB_B * SEQP)   // 10240 rows

typedef __attribute__((ext_vector_type(16))) __bf16          bf16x16;
typedef __attribute__((ext_vector_type(8)))  float           f32x8;
typedef __attribute__((ext_vector_type(8)))  unsigned short  u16x8;

union FragU { bf16x16 v; u16x8 h[2]; };

static __device__ __forceinline__ unsigned short f2bf(float f) {
  unsigned u = __builtin_bit_cast(unsigned, f);
  unsigned r = u + 0x7FFFu + ((u >> 16) & 1u);   // round-to-nearest-even
  return (unsigned short)(r >> 16);
}

#define WMMA_BF16(a, b, c) \
  __builtin_amdgcn_wmma_f32_16x16x32_bf16(false, (a), false, (b), (short)0, (c), false, false)

static __device__ __forceinline__ float blk_sum(float v, float* red) {
  const int t = threadIdx.x;
  red[t] = v; __syncthreads();
  #pragma unroll
  for (int s = 128; s > 0; s >>= 1) {
    if (t < s) red[t] += red[t + s];
    __syncthreads();
  }
  float r = red[0]; __syncthreads();
  return r;
}

// ---------------------------------------------------------------------------
// Generic GEMM: C[M,N] = act(A[M,K] * W[K,N] + bias).  ACT: 0=none, 1=gelu.
// Block tile 128x64, 8 waves (4x2), each wave 32x32 (2x2 WMMA tiles), K step 32.
// ---------------------------------------------------------------------------
template <int ACT>
__global__ __launch_bounds__(256) void gemm_wmma_kernel(
    const float* __restrict__ A, const float* __restrict__ W,
    const float* __restrict__ bias, float* __restrict__ C,
    int M, int N, int K)
{
  constexpr int SA = 40, SB = 40;     // padded LDS strides (ushorts)
  __shared__ unsigned short lsA[128 * SA];
  __shared__ unsigned short lsB[64 * SB];

  const int tid  = threadIdx.x;
  const int lane = tid & 31, hi = lane >> 4, lm = lane & 15;
  const int wv   = tid >> 5, wm = wv & 3, wn = wv >> 2;
  const int m0   = blockIdx.y * 128, n0 = blockIdx.x * 64;

  f32x8 acc[2][2] = {};

  const int ar = tid >> 1,  ac = (tid & 1) * 16;   // A tile loader: row, col base
  const int bk = tid >> 3,  bn = (tid & 7) * 8;    // B tile loader: k-row, n base

  for (int k0 = 0; k0 < K; k0 += 32) {
    // ---- stage A tile (128x32 fp32 -> bf16, row-major) ----
    {
      const int gr = m0 + ar;
      float4 v0{}, v1{}, v2{}, v3{};
      if (gr < M) {
        const float* src = A + (size_t)gr * K + k0 + ac;
        v0 = *(const float4*)(src);      v1 = *(const float4*)(src + 4);
        v2 = *(const float4*)(src + 8);  v3 = *(const float4*)(src + 12);
      }
      unsigned short* d = &lsA[ar * SA + ac];
      d[0] = f2bf(v0.x);  d[1] = f2bf(v0.y);  d[2]  = f2bf(v0.z);  d[3]  = f2bf(v0.w);
      d[4] = f2bf(v1.x);  d[5] = f2bf(v1.y);  d[6]  = f2bf(v1.z);  d[7]  = f2bf(v1.w);
      d[8] = f2bf(v2.x);  d[9] = f2bf(v2.y);  d[10] = f2bf(v2.z);  d[11] = f2bf(v2.w);
      d[12]= f2bf(v3.x);  d[13]= f2bf(v3.y);  d[14] = f2bf(v3.z);  d[15] = f2bf(v3.w);
    }
    // ---- stage B tile transposed: lsB[n][k] = W[k0+k][n0+n] ----
    {
      const float* src = W + (size_t)(k0 + bk) * N + n0 + bn;
      float4 w0 = *(const float4*)src, w1 = *(const float4*)(src + 4);
      unsigned short t[8] = { f2bf(w0.x), f2bf(w0.y), f2bf(w0.z), f2bf(w0.w),
                              f2bf(w1.x), f2bf(w1.y), f2bf(w1.z), f2bf(w1.w) };
      #pragma unroll
      for (int j = 0; j < 8; ++j) lsB[(bn + j) * SB + bk] = t[j];
    }
    // prefetch next K tile (global_prefetch_b8)
    if (k0 + 32 < K) {
      if (m0 + ar < M) __builtin_prefetch(A + (size_t)(m0 + ar) * K + k0 + 32 + ac, 0, 1);
      __builtin_prefetch(W + (size_t)(k0 + 32 + bk) * N + n0 + bn, 0, 1);
    }
    __syncthreads();

    // ---- fragments + 4 WMMAs per wave ----
    FragU af[2], bf[2];
    #pragma unroll
    for (int mt = 0; mt < 2; ++mt) {
      const unsigned short* p = &lsA[(wm * 32 + mt * 16 + lm) * SA];
      af[mt].h[0] = *(const u16x8*)(p + hi * 8);
      af[mt].h[1] = *(const u16x8*)(p + 16 + hi * 8);
    }
    #pragma unroll
    for (int nt = 0; nt < 2; ++nt) {
      const unsigned short* p = &lsB[(wn * 32 + nt * 16 + lm) * SB + hi * 8];
      bf[nt].h[0] = *(const u16x8*)(p);
      bf[nt].h[1] = *(const u16x8*)(p + 16);
    }
    #pragma unroll
    for (int mt = 0; mt < 2; ++mt)
      #pragma unroll
      for (int nt = 0; nt < 2; ++nt)
        acc[mt][nt] = WMMA_BF16(af[mt].v, bf[nt].v, acc[mt][nt]);
    __syncthreads();
  }

  // ---- epilogue: bias (+gelu) -> C ----
  #pragma unroll
  for (int mt = 0; mt < 2; ++mt)
    #pragma unroll
    for (int nt = 0; nt < 2; ++nt)
      #pragma unroll
      for (int r = 0; r < 8; ++r) {
        const int m = m0 + wm * 32 + mt * 16 + hi * 8 + r;
        if (m < M) {
          const int n = n0 + wn * 32 + nt * 16 + lm;
          float v = acc[mt][nt][r];
          if (bias) v += bias[n];
          if (ACT == 1) v = 0.5f * v * (1.0f + erff(v * 0.70710678118654752f));
          C[(size_t)m * N + n] = v;
        }
      }
}

// ---------------------------------------------------------------------------
// Attention: one block per (b, h, q-tile of 32). 4 waves / 128 threads.
// Stages K (320x64) and V^T (64x320) as bf16 in LDS, QK^T + causal softmax + PV
// entirely with WMMA. ~157KB LDS (CDNA5 allows 320KB per workgroup).
// ---------------------------------------------------------------------------
__global__ __launch_bounds__(128) void attn_kernel(
    const float* __restrict__ QKV, float* __restrict__ O)
{
  constexpr int SK = 72, SV = 328, SQ = 72, SS = 336, SP = 328;
  __shared__ unsigned short lsK[SEQP * SK];   // K   [seqk][dh]
  __shared__ unsigned short lsV[DHD * SV];    // V^T [dh][seqk]
  __shared__ unsigned short lsQ[32 * SQ];     // Q tile [32][64] (pre-scaled)
  __shared__ float          lsS[32 * SS];     // scores [32][320]
  __shared__ unsigned short lsP[32 * SP];     // probs bf16 [32][320]
  __shared__ float redM[128], redS[128];

  const int qt = blockIdx.x, h = blockIdx.y, b = blockIdx.z;
  const int tid  = threadIdx.x;
  const int lane = tid & 31, hi = lane >> 4, lm = lane & 15;
  const int wv   = tid >> 5;
  const float* qkvb = QKV + (size_t)b * SEQP * (3 * DM);
  const int lr = tid >> 2, lc = (tid & 3) * 16;

  // Q tile (scaled by 1/sqrt(DH) = 1/8)
  {
    const float* src = qkvb + (size_t)(qt * 32 + lr) * (3 * DM) + h * DHD + lc;
    unsigned short* d = &lsQ[lr * SQ + lc];
    #pragma unroll
    for (int j = 0; j < 16; ++j) d[j] = f2bf(src[j] * 0.125f);
  }
  // K rows and transposed V
  for (int rr = 0; rr < SEQP; rr += 32) {
    const int r = rr + lr;
    const float* sk = qkvb + (size_t)r * (3 * DM) + DM + h * DHD + lc;
    unsigned short* dk = &lsK[r * SK + lc];
    #pragma unroll
    for (int j = 0; j < 16; ++j) dk[j] = f2bf(sk[j]);
    const float* sv = qkvb + (size_t)r * (3 * DM) + 2 * DM + h * DHD + lc;
    #pragma unroll
    for (int j = 0; j < 16; ++j) lsV[(lc + j) * SV + r] = f2bf(sv[j]);
  }
  __syncthreads();

  // Preload Q fragments (2 m-tiles x 2 k-chunks)
  FragU qf[2][2];
  #pragma unroll
  for (int mt = 0; mt < 2; ++mt)
    #pragma unroll
    for (int kc = 0; kc < 2; ++kc) {
      const unsigned short* p = &lsQ[(mt * 16 + lm) * SQ + kc * 32];
      qf[mt][kc].h[0] = *(const u16x8*)(p + hi * 8);
      qf[mt][kc].h[1] = *(const u16x8*)(p + 16 + hi * 8);
    }

  // scores S = (Q/8) K^T : wave wv covers k-seq tiles {wv, wv+4, ...}
  for (int nt = wv; nt < SEQP / 16; nt += 4) {
    f32x8 sa[2] = {};
    #pragma unroll
    for (int kc = 0; kc < 2; ++kc) {
      FragU kf;
      const unsigned short* p = &lsK[(nt * 16 + lm) * SK + kc * 32 + hi * 8];
      kf.h[0] = *(const u16x8*)p;  kf.h[1] = *(const u16x8*)(p + 16);
      sa[0] = WMMA_BF16(qf[0][kc].v, kf.v, sa[0]);
      sa[1] = WMMA_BF16(qf[1][kc].v, kf.v, sa[1]);
    }
    #pragma unroll
    for (int mt = 0; mt < 2; ++mt)
      #pragma unroll
      for (int r = 0; r < 8; ++r)
        lsS[(mt * 16 + hi * 8 + r) * SS + nt * 16 + lm] = sa[mt][r];
  }
  __syncthreads();

  // causal mask (reference block mask == k>q) + padding mask
  for (int i = tid; i < 32 * SEQP; i += 128) {
    const int r = i / SEQP, c2 = i - r * SEQP;
    const int qg = qt * 32 + r;
    float s = lsS[r * SS + c2];
    if (c2 > qg || c2 >= SEQR) s = -1e9f;
    lsS[r * SS + c2] = s;
  }
  __syncthreads();

  // softmax: thread (row = tid&31, segment = tid>>5 of 80 cols)
  {
    const int sr = tid & 31, sg = tid >> 5;
    float mx = -3.0e38f;
    for (int c2 = sg * 80; c2 < sg * 80 + 80; ++c2) mx = fmaxf(mx, lsS[sr * SS + c2]);
    redM[sg * 32 + sr] = mx; __syncthreads();
    const float rmx = fmaxf(fmaxf(redM[sr], redM[32 + sr]),
                            fmaxf(redM[64 + sr], redM[96 + sr]));
    float sum = 0.f;
    for (int c2 = sg * 80; c2 < sg * 80 + 80; ++c2) {
      const float e = __expf(lsS[sr * SS + c2] - rmx);
      lsS[sr * SS + c2] = e; sum += e;
    }
    redS[sg * 32 + sr] = sum; __syncthreads();
    const float inv = 1.f / (redS[sr] + redS[32 + sr] + redS[64 + sr] + redS[96 + sr]);
    for (int c2 = sg * 80; c2 < sg * 80 + 80; ++c2)
      lsP[sr * SP + c2] = f2bf(lsS[sr * SS + c2] * inv);
  }
  __syncthreads();

  // O = P V : wave wv owns dh columns [wv*16, wv*16+16)
  f32x8 oa[2] = {};
  for (int kc = 0; kc < SEQP / 32; ++kc) {
    FragU pf[2], vf;
    #pragma unroll
    for (int mt = 0; mt < 2; ++mt) {
      const unsigned short* p = &lsP[(mt * 16 + lm) * SP + kc * 32];
      pf[mt].h[0] = *(const u16x8*)(p + hi * 8);
      pf[mt].h[1] = *(const u16x8*)(p + 16 + hi * 8);
    }
    const unsigned short* pv = &lsV[(wv * 16 + lm) * SV + kc * 32 + hi * 8];
    vf.h[0] = *(const u16x8*)pv;  vf.h[1] = *(const u16x8*)(pv + 16);
    oa[0] = WMMA_BF16(pf[0].v, vf.v, oa[0]);
    oa[1] = WMMA_BF16(pf[1].v, vf.v, oa[1]);
  }
  #pragma unroll
  for (int mt = 0; mt < 2; ++mt)
    #pragma unroll
    for (int r = 0; r < 8; ++r)
      O[(size_t)(b * SEQP + qt * 32 + mt * 16 + hi * 8 + r) * DM + h * DHD + wv * 16 + lm] =
          oa[mt][r];
}

// ---------------------------------------------------------------------------
// x = LayerNorm(x + t) * s + b  (in place); padded rows forced to 0.
// ---------------------------------------------------------------------------
__global__ __launch_bounds__(256) void add_ln_kernel(
    float* __restrict__ X, const float* __restrict__ T,
    const float* __restrict__ sc, const float* __restrict__ bc)
{
  const int row = blockIdx.x, sp = row % SEQP;
  const int c = threadIdx.x * 4;
  float* xr = X + (size_t)row * DM;
  if (sp >= SEQR) { xr[c] = xr[c+1] = xr[c+2] = xr[c+3] = 0.f; return; }
  __shared__ float red[256];
  const float4 xv = *(const float4*)(xr + c);
  const float4 tv = *(const float4*)(T + (size_t)row * DM + c);
  float t0 = xv.x + tv.x, t1 = xv.y + tv.y, t2 = xv.z + tv.z, t3 = xv.w + tv.w;
  const float mu = blk_sum(t0 + t1 + t2 + t3, red) * (1.f / DM);
  const float d0 = t0 - mu, d1 = t1 - mu, d2 = t2 - mu, d3 = t3 - mu;
  const float var = blk_sum(d0*d0 + d1*d1 + d2*d2 + d3*d3, red) * (1.f / DM);
  const float inv = rsqrtf(var + 1e-5f);
  const float4 sv = *(const float4*)(sc + c), bv = *(const float4*)(bc + c);
  xr[c+0] = d0 * inv * sv.x + bv.x;
  xr[c+1] = d1 * inv * sv.y + bv.y;
  xr[c+2] = d2 * inv * sv.z + bv.z;
  xr[c+3] = d3 * inv * sv.w + bv.w;
}

// ---------------------------------------------------------------------------
// Assemble input sequence X[b, s, :] = token_embed + pos; padded rows -> 0.
// img rows come from the IMG GEMM result (bias already added there).
// ---------------------------------------------------------------------------
__global__ __launch_bounds__(256) void assemble_kernel(
    const float* __restrict__ img, const float* __restrict__ states,
    const float* __restrict__ lang, const float* __restrict__ stW,
    const float* __restrict__ stB, const float* __restrict__ lgW,
    const float* __restrict__ lgB, const float* __restrict__ goalq,
    const float* __restrict__ bwdq, const float* __restrict__ fwdq,
    const float* __restrict__ pos, float* __restrict__ X)
{
  const int row = blockIdx.x, b = row / SEQP, s = row - b * SEQP;
  const int c = threadIdx.x * 4;
  float* dst = X + (size_t)row * DM + c;
  if (s >= SEQR) { dst[0] = dst[1] = dst[2] = dst[3] = 0.f; return; }
  float v0, v1, v2, v3;
  if (s == 0) {
    v0 = lgB[c]; v1 = lgB[c+1]; v2 = lgB[c+2]; v3 = lgB[c+3];
    const float* lf = lang + b * DLG;
    for (int k = 0; k < DLG; ++k) {
      const float a = lf[k];
      const float* w = lgW + (size_t)k * DM + c;
      v0 += a * w[0]; v1 += a * w[1]; v2 += a * w[2]; v3 += a * w[3];
    }
  } else if (s <= 2) {
    v0 = stB[c]; v1 = stB[c+1]; v2 = stB[c+2]; v3 = stB[c+3];
    const float* sf = states + (b * 2 + (s - 1)) * 8;
    #pragma unroll
    for (int k = 0; k < 8; ++k) {
      const float a = sf[k];
      const float* w = stW + (size_t)k * DM + c;
      v0 += a * w[0]; v1 += a * w[1]; v2 += a * w[2]; v3 += a * w[3];
    }
  } else if (s < GS0) {
    const float* ir = img + (size_t)(b * 256 + (s - 3)) * DM + c;
    v0 = ir[0]; v1 = ir[1]; v2 = ir[2]; v3 = ir[3];
  } else {
    const float* q;
    if      (s < GS0 + 16) q = goalq + (size_t)(s - GS0) * DM + c;
    else if (s < GS0 + 32) q = bwdq  + (size_t)(s - GS0 - 16) * DM + c;
    else                   q = fwdq  + (size_t)(s - GS0 - 32) * DM + c;
    v0 = q[0]; v1 = q[1]; v2 = q[2]; v3 = q[3];
  }
  const float* pe = pos + (size_t)s * DM + c;
  dst[0] = v0 + pe[0]; dst[1] = v1 + pe[1]; dst[2] = v2 + pe[2]; dst[3] = v3 + pe[3];
}

// ---------------------------------------------------------------------------
// RMSNorm + mean-pool the goal / bwd / fwd token groups (16 rows each).
// pool[b, g, :] = mean_r( x_r * rsqrt(mean(x_r^2)+1e-6) ) * final_rms
// ---------------------------------------------------------------------------
__global__ __launch_bounds__(256) void rms_pool_kernel(
    const float* __restrict__ X, const float* __restrict__ frms,
    float* __restrict__ pool)
{
  const int b = blockIdx.x, g = blockIdx.y;
  __shared__ float red[256];
  const int c = threadIdx.x * 4;
  float a0 = 0.f, a1 = 0.f, a2 = 0.f, a3 = 0.f;
  for (int rr = 0; rr < 16; ++rr) {
    const float* xr = X + (size_t)(b * SEQP + GS0 + g * 16 + rr) * DM;
    const float4 xv = *(const float4*)(xr + c);
    const float ss = blk_sum(xv.x*xv.x + xv.y*xv.y + xv.z*xv.z + xv.w*xv.w, red);
    const float inv = rsqrtf(ss * (1.f / DM) + 1e-6f);
    a0 += xv.x * inv; a1 += xv.y * inv; a2 += xv.z * inv; a3 += xv.w * inv;
  }
  const float4 fr = *(const float4*)(frms + c);
  float* op = pool + (size_t)(b * 3 + g) * DM + c;
  op[0] = a0 * fr.x * (1.f / 16.f);
  op[1] = a1 * fr.y * (1.f / 16.f);
  op[2] = a2 * fr.z * (1.f / 16.f);
  op[3] = a3 * fr.w * (1.f / 16.f);
}

// ---------------------------------------------------------------------------
// Tiny head projections (total 33M MACs, negligible): d_out = latent||bwd||fwd.
// ---------------------------------------------------------------------------
__global__ __launch_bounds__(256) void heads_kernel(
    const float* __restrict__ pool,
    const float* __restrict__ gW, const float* __restrict__ gb,
    const float* __restrict__ bW, const float* __restrict__ bb,
    const float* __restrict__ fW, const float* __restrict__ fb,
    float* __restrict__ out)
{
  const int idx = blockIdx.x * 256 + threadIdx.x;   // 0..32767
  const float* prow; const float* W; float acc; int n, ldw;
  if (idx < 24576) {
    const int b = idx / 768; n = idx - b * 768;
    prow = pool + (size_t)(b * 3 + 0) * DM; W = gW; ldw = 768; acc = gb[n];
  } else if (idx < 28672) {
    const int j = idx - 24576; const int b = j / 128; n = j - b * 128;
    prow = pool + (size_t)(b * 3 + 1) * DM; W = bW; ldw = 128; acc = bb[n];
  } else {
    const int j = idx - 28672; const int b = j / 128; n = j - b * 128;
    prow = pool + (size_t)(b * 3 + 2) * DM; W = fW; ldw = 128; acc = fb[n];
  }
  for (int k = 0; k < DM; ++k) acc += prow[k] * W[(size_t)k * ldw + n];
  out[idx] = acc;
}

// ---------------------------------------------------------------------------
// Host launcher. Inputs in setup_inputs() insertion order (recursive).
// Workspace layout (floats): X(10.49M) | T(10.49M) | RA(41.94M) | POOL(98K)
//   RA = QKV(31.46M)+O(10.49M) during attention; reused as H(41.94M) for FFN.
// Total ~240.5 MiB.
// ---------------------------------------------------------------------------
extern "C" void kernel_launch(void* const* d_in, const int* in_sizes, int n_in,
                              void* d_out, int out_size, void* d_ws, size_t ws_size,
                              hipStream_t stream)
{
  (void)in_sizes; (void)n_in; (void)out_size; (void)ws_size;
  const float* image_tokens = (const float*)d_in[0];
  const float* states    = (const float*)d_in[1];
  const float* lang      = (const float*)d_in[2];
  const float* state_W   = (const float*)d_in[3];
  const float* state_b   = (const float*)d_in[4];
  const float* lang_W    = (const float*)d_in[5];
  const float* lang_b    = (const float*)d_in[6];
  const float* img_W     = (const float*)d_in[7];
  const float* img_b     = (const float*)d_in[8];
  const float* goal_q    = (const float*)d_in[9];
  const float* bwd_q     = (const float*)d_in[10];
  const float* fwd_q     = (const float*)d_in[11];
  const float* pos       = (const float*)d_in[12];
  const float* Wqkv      = (const float*)d_in[13];
  const float* bqkv      = (const float*)d_in[14];
  const float* Wo        = (const float*)d_in[15];
  const float* bo        = (const float*)d_in[16];
  const float* ln1_s     = (const float*)d_in[17];
  const float* ln1_b     = (const float*)d_in[18];
  const float* W1        = (const float*)d_in[19];
  const float* b1        = (const float*)d_in[20];
  const float* W2        = (const float*)d_in[21];
  const float* b2        = (const float*)d_in[22];
  const float* ln2_s     = (const float*)d_in[23];
  const float* ln2_b     = (const float*)d_in[24];
  const float* final_rms = (const float*)d_in[25];
  const float* goal_Wp   = (const float*)d_in[26];
  const float* goal_bp   = (const float*)d_in[27];
  const float* bwd_Wp    = (const float*)d_in[28];
  const float* bwd_bp    = (const float*)d_in[29];
  const float* fwd_Wp    = (const float*)d_in[30];
  const float* fwd_bp    = (const float*)d_in[31];

  float* X    = (float*)d_ws;
  float* T    = X + (size_t)MTOT * DM;
  float* RA   = T + (size_t)MTOT * DM;
  float* QKV  = RA;
  float* Obuf = RA + (size_t)MTOT * 3 * DM;
  float* Hbuf = RA;
  float* POOL = RA + (size_t)MTOT * DFF_;

  // image token projection: [B*256, 768] x [768, 1024] + img_b -> T
  gemm_wmma_kernel<0><<<dim3(DM / 64, (NB_B * 256) / 128), 256, 0, stream>>>(
      image_tokens, img_W, img_b, T, NB_B * 256, DM, DIMG);
  // assemble sequence (+pos, queries, lang/state embeds) -> X
  assemble_kernel<<<MTOT, 256, 0, stream>>>(T, states, lang, state_W, state_b,
      lang_W, lang_b, goal_q, bwd_q, fwd_q, pos, X);

  for (int L = 0; L < NLAY; ++L) {
    gemm_wmma_kernel<0><<<dim3(3 * DM / 64, MTOT / 128), 256, 0, stream>>>(
        X, Wqkv + (size_t)L * DM * 3 * DM, bqkv + (size_t)L * 3 * DM, QKV,
        MTOT, 3 * DM, DM);
    attn_kernel<<<dim3(SEQP / 32, NH, NB_B), 128, 0, stream>>>(QKV, Obuf);
    gemm_wmma_kernel<0><<<dim3(DM / 64, MTOT / 128), 256, 0, stream>>>(
        Obuf, Wo + (size_t)L * DM * DM, bo + (size_t)L * DM, T, MTOT, DM, DM);
    add_ln_kernel<<<MTOT, 256, 0, stream>>>(X, T, ln1_s + (size_t)L * DM,
                                            ln1_b + (size_t)L * DM);
    gemm_wmma_kernel<1><<<dim3(DFF_ / 64, MTOT / 128), 256, 0, stream>>>(
        X, W1 + (size_t)L * DM * DFF_, b1 + (size_t)L * DFF_, Hbuf,
        MTOT, DFF_, DM);
    gemm_wmma_kernel<0><<<dim3(DM / 64, MTOT / 128), 256, 0, stream>>>(
        Hbuf, W2 + (size_t)L * DFF_ * DM, b2 + (size_t)L * DM, T, MTOT, DM, DFF_);
    add_ln_kernel<<<MTOT, 256, 0, stream>>>(X, T, ln2_s + (size_t)L * DM,
                                            ln2_b + (size_t)L * DM);
  }

  rms_pool_kernel<<<dim3(NB_B, 3), 256, 0, stream>>>(X, final_rms, POOL);
  heads_kernel<<<128, 256, 0, stream>>>(POOL, goal_Wp, goal_bp, bwd_Wp, bwd_bp,
                                        fwd_Wp, fwd_bp, (float*)d_out);
}
